// Block_38543036514772
// MI455X (gfx1250) — matrix operand order repaired and checked by reading, not compiled, not run on hardware
//
#include <hip/hip_runtime.h>
#include <hip/hip_bf16.h>
#include <math.h>

typedef unsigned short u16;
typedef unsigned int   u32;
typedef __attribute__((ext_vector_type(16))) __bf16 v16bf;
typedef __attribute__((ext_vector_type(8)))  float  v8f;

#define WMMA_BF16(a,b,c) __builtin_amdgcn_wmma_f32_16x16x32_bf16(false,(a),false,(b),(short)0,(c),false,false)

#define DIMC   768
#define HEADS  12
#define HD     64
#define NT     4816896      /* 6272 * 768 tokens*dim  */
#define TOK    6272

__device__ __forceinline__ u16 f2bf(float f){
  u32 u = __float_as_uint(f);
  u32 r = u + 0x7fffu + ((u >> 16) & 1u);
  return (u16)(r >> 16);
}
__device__ __forceinline__ float bf2f(u16 h){ return __uint_as_float(((u32)h) << 16); }

// A-fragment (16xK tile of row-major [rows,K] bf16): lane holds row (passed in),
// K offsets {c..c+7, c+16..c+23}, c = 8*(lane>>4).
__device__ __forceinline__ v16bf load_a_bf(const u16* base, int ld, int row, int k0, int lane){
  int c = k0 + ((lane >> 4) << 3);
  const u16* p = base + (size_t)row * ld + c;
  union { uint4 u[2]; v16bf v; } x;
  x.u[0] = *(const uint4*)p;
  x.u[1] = *(const uint4*)(p + 16);
  return x.v;
}
// B-fragment (Kx16, sourced from row-major [cols,K] bf16, i.e. W so B=W^T):
// lane holds column-row (passed in), K offsets {c..c+7, c+16..c+23}, c = 8*(lane>>4).
__device__ __forceinline__ v16bf load_b_bf(const u16* base, int ld, int row, int k0, int lane){
  int c = k0 + ((lane >> 4) << 3);
  const u16* p = base + (size_t)row * ld + c;
  union { uint4 u[2]; v16bf v; } x;
  x.u[0] = *(const uint4*)p;
  x.u[1] = *(const uint4*)(p + 16);
  return x.v;
}

__device__ __forceinline__ float gelu_erf(float x){
  return 0.5f * x * (1.0f + erff(x * 0.70710678118654752f));
}

// ------- GEMM (small-N path, N%64==0): per-wave 16x64, block 128x64 -------
__global__ __launch_bounds__(256) void gemm_bf16(
    const u16* __restrict__ A, const u16* __restrict__ W, const float* __restrict__ bias,
    float* __restrict__ outF, u16* __restrict__ outB, int M, int N, int K, int act)
{
  int lane = threadIdx.x & 31, wave = threadIdx.x >> 5;
  int m0 = blockIdx.y * 128 + wave * 16;
  int n0 = blockIdx.x * 64;
  int col = lane & 15, half = lane >> 4;
  int arow = m0 + col;
  v8f acc[4] = {};
  for (int k0 = 0; k0 < K; k0 += 32) {
    v16bf a = load_a_bf(A, K, arow, k0, lane);
#pragma unroll
    for (int t = 0; t < 4; ++t) {
      v16bf b = load_b_bf(W, K, n0 + t * 16 + col, k0, lane);
      acc[t] = WMMA_BF16(a, b, acc[t]);
    }
  }
#pragma unroll
  for (int t = 0; t < 4; ++t) {
    int n = n0 + t * 16 + col;
    float bv = bias ? bias[n] : 0.0f;
#pragma unroll
    for (int r = 0; r < 8; ++r) {
      int m = m0 + r + 8 * half;
      float x = acc[t][r] + bv;
      if (act == 1) x = gelu_erf(x);
      size_t idx = (size_t)m * N + n;
      if (outF) outF[idx] = x;
      if (outB) outB[idx] = f2bf(x);
    }
  }
}

// ------- GEMM (main path, N%128==0): per-wave 32x64, block 128x128,
//         software-pipelined k-loop (prefetch k+32 fragments before math) -------
__global__ __launch_bounds__(256) void gemm_bf16_big(
    const u16* __restrict__ A, const u16* __restrict__ W, const float* __restrict__ bias,
    float* __restrict__ outF, u16* __restrict__ outB, int M, int N, int K, int act)
{
  int lane = threadIdx.x & 31, wave = threadIdx.x >> 5;
  int mq = wave & 3, nq = wave >> 2;
  int m0 = blockIdx.y * 128 + mq * 32;
  int n0 = blockIdx.x * 128 + nq * 64;
  int col = lane & 15, half = lane >> 4;
  int arow0 = m0 + col, arow1 = m0 + 16 + col;

  v8f acc[2][4] = {};
  v16bf a0 = load_a_bf(A, K, arow0, 0, lane);
  v16bf a1 = load_a_bf(A, K, arow1, 0, lane);
  v16bf b0 = load_b_bf(W, K, n0 +  0 + col, 0, lane);
  v16bf b1 = load_b_bf(W, K, n0 + 16 + col, 0, lane);
  v16bf b2 = load_b_bf(W, K, n0 + 32 + col, 0, lane);
  v16bf b3 = load_b_bf(W, K, n0 + 48 + col, 0, lane);

  for (int k0 = 0; k0 < K - 32; k0 += 32) {
    int kn = k0 + 32;
    // prefetch next k-step into registers
    v16bf a0n = load_a_bf(A, K, arow0, kn, lane);
    v16bf a1n = load_a_bf(A, K, arow1, kn, lane);
    v16bf b0n = load_b_bf(W, K, n0 +  0 + col, kn, lane);
    v16bf b1n = load_b_bf(W, K, n0 + 16 + col, kn, lane);
    v16bf b2n = load_b_bf(W, K, n0 + 32 + col, kn, lane);
    v16bf b3n = load_b_bf(W, K, n0 + 48 + col, kn, lane);
    // 8 independent WMMAs on current fragments
    acc[0][0] = WMMA_BF16(a0, b0, acc[0][0]);
    acc[0][1] = WMMA_BF16(a0, b1, acc[0][1]);
    acc[0][2] = WMMA_BF16(a0, b2, acc[0][2]);
    acc[0][3] = WMMA_BF16(a0, b3, acc[0][3]);
    acc[1][0] = WMMA_BF16(a1, b0, acc[1][0]);
    acc[1][1] = WMMA_BF16(a1, b1, acc[1][1]);
    acc[1][2] = WMMA_BF16(a1, b2, acc[1][2]);
    acc[1][3] = WMMA_BF16(a1, b3, acc[1][3]);
    a0 = a0n; a1 = a1n; b0 = b0n; b1 = b1n; b2 = b2n; b3 = b3n;
  }
  acc[0][0] = WMMA_BF16(a0, b0, acc[0][0]);
  acc[0][1] = WMMA_BF16(a0, b1, acc[0][1]);
  acc[0][2] = WMMA_BF16(a0, b2, acc[0][2]);
  acc[0][3] = WMMA_BF16(a0, b3, acc[0][3]);
  acc[1][0] = WMMA_BF16(a1, b0, acc[1][0]);
  acc[1][1] = WMMA_BF16(a1, b1, acc[1][1]);
  acc[1][2] = WMMA_BF16(a1, b2, acc[1][2]);
  acc[1][3] = WMMA_BF16(a1, b3, acc[1][3]);

#pragma unroll
  for (int s = 0; s < 2; ++s) {
#pragma unroll
    for (int t = 0; t < 4; ++t) {
      int n = n0 + t * 16 + col;
      float bv = bias ? bias[n] : 0.0f;
#pragma unroll
      for (int r = 0; r < 8; ++r) {
        int m = m0 + s * 16 + r + 8 * half;
        float x = acc[s][t][r] + bv;
        if (act == 1) x = gelu_erf(x);
        size_t idx = (size_t)m * N + n;
        if (outF) outF[idx] = x;
        if (outB) outB[idx] = f2bf(x);
      }
    }
  }
}

// ---------------- LayerNorm over 768, bf16 out ----------------
__global__ __launch_bounds__(256) void ln_kernel(const float* __restrict__ x, const float* __restrict__ g,
                                                 const float* __restrict__ b, u16* __restrict__ out)
{
  __shared__ float red[256];
  int tid = threadIdx.x;
  size_t base = (size_t)blockIdx.x * DIMC;
  float v0 = x[base + tid], v1 = x[base + tid + 256], v2 = x[base + tid + 512];
  red[tid] = v0 + v1 + v2; __syncthreads();
  for (int s = 128; s > 0; s >>= 1) { if (tid < s) red[tid] += red[tid + s]; __syncthreads(); }
  float mean = red[0] * (1.0f / 768.0f);
  __syncthreads();
  float d0 = v0 - mean, d1 = v1 - mean, d2 = v2 - mean;
  red[tid] = d0 * d0 + d1 * d1 + d2 * d2; __syncthreads();
  for (int s = 128; s > 0; s >>= 1) { if (tid < s) red[tid] += red[tid + s]; __syncthreads(); }
  float rs = rsqrtf(red[0] * (1.0f / 768.0f) + 1e-5f);
  out[base + tid]       = f2bf(d0 * rs * g[tid]       + b[tid]);
  out[base + tid + 256] = f2bf(d1 * rs * g[tid + 256] + b[tid + 256]);
  out[base + tid + 512] = f2bf(d2 * rs * g[tid + 512] + b[tid + 512]);
}

// ---------------- Flash attention: one wave per 16-query tile, HD=64 ----------------
__global__ __launch_bounds__(32) void attn_flash(
    const u16* __restrict__ Qb, const u16* __restrict__ Kb, const u16* __restrict__ Vb,
    u16* __restrict__ Ob, int qs_, int kvs_, int os_, int q_len, int k_len)
{
  __shared__ __align__(16) float s_p[16 * 32];
  __shared__ __align__(16) u16   s_v[64 * 32];
  int lane = threadIdx.x;
  int pair = blockIdx.y; int g = pair / HEADS, h = pair - g * HEADS;
  const u16* q = Qb + (size_t)g * q_len * qs_  + h * HD;
  const u16* k = Kb + (size_t)g * k_len * kvs_ + h * HD;
  const u16* v = Vb + (size_t)g * k_len * kvs_ + h * HD;
  u16*       o = Ob + (size_t)g * q_len * os_  + h * HD;
  int q0 = blockIdx.x * 16;
  int col = lane & 15, half = lane >> 4;
  int qrow = q0 + col; if (qrow >= q_len) qrow = q_len - 1;
  v16bf qa0 = load_a_bf(q, qs_, qrow, 0,  lane);
  v16bf qa1 = load_a_bf(q, qs_, qrow, 32, lane);
  v8f oacc[4] = {};
  float mrow[8], lrow[8];
#pragma unroll
  for (int r = 0; r < 8; ++r) { mrow[r] = -1e30f; lrow[r] = 0.0f; }

  for (int kt = 0; kt < k_len; kt += 32) {
    v8f s0 = {}, s1 = {};
    int kr0 = kt + col;      if (kr0 >= k_len) kr0 = k_len - 1;
    int kr1 = kt + 16 + col; if (kr1 >= k_len) kr1 = k_len - 1;
    v16bf b00 = load_b_bf(k, kvs_, kr0, 0,  lane);
    v16bf b01 = load_b_bf(k, kvs_, kr0, 32, lane);
    v16bf b10 = load_b_bf(k, kvs_, kr1, 0,  lane);
    v16bf b11 = load_b_bf(k, kvs_, kr1, 32, lane);
    s0 = WMMA_BF16(qa0, b00, s0); s0 = WMMA_BF16(qa1, b01, s0);
    s1 = WMMA_BF16(qa0, b10, s1); s1 = WMMA_BF16(qa1, b11, s1);
    bool inv0 = (kt + col)      >= k_len;
    bool inv1 = (kt + 16 + col) >= k_len;
#pragma unroll
    for (int r = 0; r < 8; ++r) {
      s0[r] = inv0 ? -1e30f : s0[r] * 0.125f;
      s1[r] = inv1 ? -1e30f : s1[r] * 0.125f;
    }
    { // stage V tile transposed: s_v[d*32 + key_local]
      int key = kt + lane;
      if (key < k_len) {
        const u16* vp = v + (size_t)key * kvs_;
#pragma unroll
        for (int j = 0; j < 8; ++j) {
          union { uint4 q4; u16 s[8]; } w; w.q4 = *(const uint4*)(vp + j * 8);
#pragma unroll
          for (int e = 0; e < 8; ++e) s_v[(j * 8 + e) * 32 + lane] = w.s[e];
        }
      } else {
#pragma unroll
        for (int j = 0; j < 64; ++j) s_v[j * 32 + lane] = (u16)0;
      }
    }
    // online softmax + write P to LDS
#pragma unroll
    for (int r = 0; r < 8; ++r) {
      float rm = fmaxf(s0[r], s1[r]);
#pragma unroll
      for (int off = 1; off < 16; off <<= 1) rm = fmaxf(rm, __shfl_xor(rm, off, 32));
      float nm = fmaxf(mrow[r], rm);
      float al = expf(mrow[r] - nm);
      float p0 = expf(s0[r] - nm);
      float p1 = expf(s1[r] - nm);
      float ps = p0 + p1;
#pragma unroll
      for (int off = 1; off < 16; off <<= 1) ps += __shfl_xor(ps, off, 32);
      lrow[r] = lrow[r] * al + ps;
      mrow[r] = nm;
      int m = r + 8 * half;
      s_p[m * 32 + col]      = p0;
      s_p[m * 32 + 16 + col] = p1;
#pragma unroll
      for (int t = 0; t < 4; ++t) oacc[t][r] *= al;
    }
    __syncthreads();
    // P a-fragment from LDS (f32 -> bf16)
    union { u16 s[16]; v16bf v; } pa;
    {
      int m = col, c = half * 8;
#pragma unroll
      for (int e = 0; e < 8; ++e) {
        pa.s[e]     = f2bf(s_p[m * 32 + c + e]);
        pa.s[8 + e] = f2bf(s_p[m * 32 + c + 16 + e]);
      }
    }
#pragma unroll
    for (int t = 0; t < 4; ++t) {
      union { uint4 u[2]; v16bf v; } vb;
      const u16* vp = &s_v[(t * 16 + col) * 32 + half * 8];
      vb.u[0] = *(const uint4*)vp;
      vb.u[1] = *(const uint4*)(vp + 16);
      oacc[t] = WMMA_BF16(pa.v, vb.v, oacc[t]);
    }
    __syncthreads();
  }
#pragma unroll
  for (int r = 0; r < 8; ++r) {
    int m = q0 + r + 8 * half;
    if (m < q_len) {
      float inv = 1.0f / lrow[r];
#pragma unroll
      for (int t = 0; t < 4; ++t)
        o[(size_t)m * os_ + t * 16 + col] = f2bf(oacc[t][r] * inv);
    }
  }
}

// ---------------- Tiny T2S attention: T=8 sequence, scalar ----------------
__global__ __launch_bounds__(64) void attn_t2s(const u16* __restrict__ Q, const u16* __restrict__ K,
                                               const u16* __restrict__ V, u16* __restrict__ O)
{
  __shared__ float qs[8][64], ks[8][64], vs[8][64], pr[8][8];
  int blk = blockIdx.x;
  int h = blk % HEADS, g = blk / HEADS;             // g = b*196 + loc
  size_t qbase = ((size_t)g * 8) * DIMC + h * HD;
  size_t kbase = ((size_t)g * 8) * 1536 + h * HD;
  int d = threadIdx.x;
  for (int r = 0; r < 8; ++r) {
    qs[r][d] = bf2f(Q[qbase + (size_t)r * DIMC + d]);
    ks[r][d] = bf2f(K[kbase + (size_t)r * 1536 + d]);
    vs[r][d] = bf2f(V[kbase + (size_t)r * 1536 + d]);
  }
  __syncthreads();
  {
    int qi = d >> 3, ki = d & 7;
    float s = 0.0f;
    for (int e = 0; e < 64; ++e) s += qs[qi][e] * ks[ki][e];
    pr[qi][ki] = s * 0.125f;
  }
  __syncthreads();
  if (d < 8) {
    float mx = -1e30f;
    for (int kk = 0; kk < 8; ++kk) mx = fmaxf(mx, pr[d][kk]);
    float sm = 0.0f;
    for (int kk = 0; kk < 8; ++kk) { float e = expf(pr[d][kk] - mx); pr[d][kk] = e; sm += e; }
    float inv = 1.0f / sm;
    for (int kk = 0; kk < 8; ++kk) pr[d][kk] *= inv;
  }
  __syncthreads();
  for (int qi = 0; qi < 8; ++qi) {
    float o = 0.0f;
    for (int kk = 0; kk < 8; ++kk) o += pr[qi][kk] * vs[kk][d];
    O[qbase + (size_t)qi * DIMC + d] = f2bf(o);
  }
}

// ---------------- elementwise helpers ----------------
__global__ void k_cvt(const float* __restrict__ s, u16* __restrict__ d, int n){
  int i = blockIdx.x * 256 + threadIdx.x; if (i < n) d[i] = f2bf(s[i]);
}
__global__ void k_qkv_bias(const float* __restrict__ qb, const float* __restrict__ vb, float* __restrict__ o){
  int i = blockIdx.x * 256 + threadIdx.x;
  if (i < 2304) o[i] = (i < 768) ? qb[i] : ((i < 1536) ? 0.0f : vb[i - 1536]);
}
__global__ void k_add3(float* __restrict__ o, const float* __restrict__ a, const float* __restrict__ b,
                       const float* __restrict__ c, float cs, int n){
  int i = blockIdx.x * 256 + threadIdx.x; if (i < n) o[i] = a[i] + b[i] + cs * c[i];
}
__global__ void k_add2(float* __restrict__ o, const float* __restrict__ a, const float* __restrict__ b, int n){
  int i = blockIdx.x * 256 + threadIdx.x; if (i < n) o[i] = a[i] + b[i];
}
// dst = bf16(src + pos[row % mod])
__global__ void k_posadd(u16* __restrict__ d, const float* __restrict__ src, const float* __restrict__ pos,
                         int mod, int n){
  int i = blockIdx.x * 256 + threadIdx.x;
  if (i < n) { int c = i % DIMC; int row = i / DIMC; d[i] = f2bf(src[i] + pos[(row % mod) * DIMC + c]); }
}
// dst rows (b,p,t) <- src rows (b,t,p), + pos[t]
__global__ void k_perm_posadd(u16* __restrict__ d, const float* __restrict__ src, const float* __restrict__ pos, int n){
  int i = blockIdx.x * 256 + threadIdx.x;
  if (i < n) {
    int c = i % DIMC; int rowp = i / DIMC;
    int t = rowp & 7; int bp = rowp >> 3; int p = bp % 196; int b = bp / 196;
    int srow = (b * 8 + t) * 196 + p;
    d[i] = f2bf(src[(size_t)srow * DIMC + c] + pos[t * DIMC + c]);
  }
}
// s_out[(b,t,hw)] = s_x + o3p[(b,hw,t)]
__global__ void k_sout(float* __restrict__ outs, const float* __restrict__ sx, const float* __restrict__ o3p, int n){
  int i = blockIdx.x * 256 + threadIdx.x;
  if (i < n) {
    int c = i % DIMC; int row = i / DIMC;
    int hw = row % 196; int bt = row / 196; int t = bt & 7; int b = bt >> 3;
    int r2 = (b * 196 + hw) * 8 + t;
    outs[i] = sx[i] + o3p[(size_t)r2 * DIMC + c];
  }
}

// ======================================================================
extern "C" void kernel_launch(void* const* d_in, const int* in_sizes, int n_in,
                              void* d_out, int out_size, void* d_ws, size_t ws_size,
                              hipStream_t stream) {
  (void)in_sizes; (void)n_in; (void)out_size; (void)ws_size;
  const float* s_x   = (const float*)d_in[0];
  const float* t_x   = (const float*)d_in[1];
  const float* g1    = (const float*)d_in[2];
  const float* bb1   = (const float*)d_in[3];
  const float* qkv_w = (const float*)d_in[4];
  const float* q_bias= (const float*)d_in[5];
  const float* v_bias= (const float*)d_in[6];
  const float* aproj_w=(const float*)d_in[7];
  const float* aproj_b=(const float*)d_in[8];
  const float* sa_w1 = (const float*)d_in[9];
  const float* sa_b1 = (const float*)d_in[10];
  const float* sa_w2 = (const float*)d_in[11];
  const float* sa_b2 = (const float*)d_in[12];
  const float* s2t_pos_cnn  = (const float*)d_in[13];
  const float* s2t_pos_vmae = (const float*)d_in[14];
  const float* s2t_qw = (const float*)d_in[15];
  const float* s2t_qb = (const float*)d_in[16];
  const float* s2t_kvw= (const float*)d_in[17];
  const float* s2t_kvb= (const float*)d_in[18];
  const float* s2t_pw = (const float*)d_in[19];
  const float* s2t_pb = (const float*)d_in[20];
  const float* t2s_pos_cnn  = (const float*)d_in[21];
  const float* t2s_pos_vmae = (const float*)d_in[22];
  const float* t2s_qw = (const float*)d_in[23];
  const float* t2s_qb = (const float*)d_in[24];
  const float* t2s_kvw= (const float*)d_in[25];
  const float* t2s_kvb= (const float*)d_in[26];
  const float* t2s_pw = (const float*)d_in[27];
  const float* t2s_pb = (const float*)d_in[28];
  const float* g2    = (const float*)d_in[29];
  const float* bb2   = (const float*)d_in[30];
  const float* fc1_w = (const float*)d_in[31];
  const float* fc1_b = (const float*)d_in[32];
  const float* fc2_w = (const float*)d_in[33];
  const float* fc2_b = (const float*)d_in[34];
  const float* ma_w1 = (const float*)d_in[35];
  const float* ma_b1 = (const float*)d_in[36];
  const float* ma_w2 = (const float*)d_in[37];
  const float* ma_b2 = (const float*)d_in[38];

  float* outS = (float*)d_out;        // [B*T, H, W, DIM]
  float* outT = outS + NT;            // [B, T*N, DIM]

  // ---- workspace bump allocator ----
  char* wsb = (char*)d_ws; size_t off = 0;
  auto B2 = [&](size_t e){ void* p = wsb + off; off = (off + e * 2 + 255) & ~(size_t)255; return (u16*)p; };
  auto F4 = [&](size_t e){ void* p = wsb + off; off = (off + e * 4 + 255) & ~(size_t)255; return (float*)p; };

  // persistent: bf16 weights + residual tensors
  u16* qkv_wb  = B2((size_t)2304 * 768);
  u16* aproj_wb= B2((size_t)768 * 768);
  u16* sa_w1b  = B2((size_t)192 * 768);
  u16* sa_w2b  = B2((size_t)768 * 192);
  u16* s2t_qwb = B2((size_t)768 * 768);
  u16* s2t_kvwb= B2((size_t)1536 * 768);
  u16* s2t_pwb = B2((size_t)768 * 768);
  u16* t2s_qwb = B2((size_t)768 * 768);
  u16* t2s_kvwb= B2((size_t)1536 * 768);
  u16* t2s_pwb = B2((size_t)768 * 768);
  u16* fc1_wb  = B2((size_t)3072 * 768);
  u16* fc2_wb  = B2((size_t)768 * 3072);
  u16* ma_w1b  = B2((size_t)192 * 768);
  u16* ma_w2b  = B2((size_t)768 * 192);
  float* biasQKV = F4(2304);
  float* T1 = F4(NT);
  float* T2 = F4(NT);
  size_t pool = off;

  const int EW = (NT + 255) / 256;
  auto CVT = [&](const float* s, u16* d, size_t n){
    k_cvt<<<dim3((unsigned)((n + 255) / 256)), 256, 0, stream>>>(s, d, (int)n);
  };
  auto GEMM = [&](const u16* A, const u16* W, const float* bias, float* oF, u16* oB,
                  int N, int K, int act){
    if ((N & 127) == 0)
      gemm_bf16_big<<<dim3((unsigned)(N / 128), 49), 256, 0, stream>>>(A, W, bias, oF, oB, TOK, N, K, act);
    else
      gemm_bf16<<<dim3((unsigned)(N / 64), 49), 256, 0, stream>>>(A, W, bias, oF, oB, TOK, N, K, act);
  };

  // weight conversions
  CVT(qkv_w,  qkv_wb,  (size_t)2304 * 768);
  CVT(aproj_w,aproj_wb,(size_t)768 * 768);
  CVT(sa_w1,  sa_w1b,  (size_t)192 * 768);
  CVT(sa_w2,  sa_w2b,  (size_t)768 * 192);
  CVT(s2t_qw, s2t_qwb, (size_t)768 * 768);
  CVT(s2t_kvw,s2t_kvwb,(size_t)1536 * 768);
  CVT(s2t_pw, s2t_pwb, (size_t)768 * 768);
  CVT(t2s_qw, t2s_qwb, (size_t)768 * 768);
  CVT(t2s_kvw,t2s_kvwb,(size_t)1536 * 768);
  CVT(t2s_pw, t2s_pwb, (size_t)768 * 768);
  CVT(fc1_w,  fc1_wb,  (size_t)3072 * 768);
  CVT(fc2_w,  fc2_wb,  (size_t)768 * 3072);
  CVT(ma_w1,  ma_w1b,  (size_t)192 * 768);
  CVT(ma_w2,  ma_w2b,  (size_t)768 * 192);
  k_qkv_bias<<<dim3(9), 256, 0, stream>>>(q_bias, v_bias, biasQKV);

  // ---- Phase 1: self-attention + adapter ----
  off = pool;
  u16*   XNb  = B2(NT);
  u16*   QKVb = B2((size_t)TOK * 2304);
  u16*   Ob   = B2(NT);
  float* APf  = F4(NT);
  u16*   APb  = B2(NT);
  u16*   D1b  = B2((size_t)TOK * 192);
  float* ADf  = F4(NT);

  ln_kernel<<<dim3(TOK), 256, 0, stream>>>(t_x, g1, bb1, XNb);
  GEMM(XNb, qkv_wb, biasQKV, nullptr, QKVb, 2304, 768, 0);
  attn_flash<<<dim3(98, 48), 32, 0, stream>>>(QKVb, QKVb + 768, QKVb + 1536, Ob,
                                              2304, 2304, 768, 1568, 1568);
  GEMM(Ob,  aproj_wb, aproj_b, APf, APb, 768, 768, 0);
  GEMM(APb, sa_w1b,   sa_b1,   nullptr, D1b, 192, 768, 1);
  GEMM(D1b, sa_w2b,   sa_b2,   ADf, nullptr, 768, 192, 0);
  k_add3<<<dim3(EW), 256, 0, stream>>>(T1, t_x, APf, ADf, 1.0f, NT);

  // ---- Phase 2: S2T cross-attention ----
  off = pool;
  u16*   TXb  = B2(NT);
  u16*   SPb  = B2(NT);
  u16*   Q2b  = B2(NT);
  u16*   KV2b = B2((size_t)TOK * 1536);
  u16*   O2b  = B2(NT);
  float* O2p  = F4(NT);

  k_posadd<<<dim3(EW), 256, 0, stream>>>(TXb, T1,  s2t_pos_vmae, 196, NT);
  k_posadd<<<dim3(EW), 256, 0, stream>>>(SPb, s_x, s2t_pos_cnn,  196, NT);
  GEMM(TXb, s2t_qwb,  s2t_qb,  nullptr, Q2b,  768, 768, 0);
  GEMM(SPb, s2t_kvwb, s2t_kvb, nullptr, KV2b, 1536, 768, 0);
  attn_flash<<<dim3(13, 384), 32, 0, stream>>>(Q2b, KV2b, KV2b + 768, O2b,
                                               768, 1536, 768, 196, 196);
  GEMM(O2b, s2t_pwb, s2t_pb, O2p, nullptr, 768, 768, 0);
  k_add2<<<dim3(EW), 256, 0, stream>>>(T2, T1, O2p, NT);

  // ---- Phase 3: T2S cross-attention ----
  off = pool;
  u16*   SP2b = B2(NT);
  u16*   TX2b = B2(NT);
  u16*   Q3b  = B2(NT);
  u16*   KV3b = B2((size_t)TOK * 1536);
  u16*   O3b  = B2(NT);
  float* O3p  = F4(NT);

  k_perm_posadd<<<dim3(EW), 256, 0, stream>>>(SP2b, s_x, t2s_pos_cnn,  NT);
  k_perm_posadd<<<dim3(EW), 256, 0, stream>>>(TX2b, T2,  t2s_pos_vmae, NT);
  GEMM(SP2b, t2s_qwb,  t2s_qb,  nullptr, Q3b,  768, 768, 0);
  GEMM(TX2b, t2s_kvwb, t2s_kvb, nullptr, KV3b, 1536, 768, 0);
  attn_t2s<<<dim3(784 * HEADS), 64, 0, stream>>>(Q3b, KV3b, KV3b + 768, O3b);
  GEMM(O3b, t2s_pwb, t2s_pb, O3p, nullptr, 768, 768, 0);
  k_sout<<<dim3(EW), 256, 0, stream>>>(outS, s_x, O3p, NT);

  // ---- Phase 4: MLP + parallel adapter ----
  off = pool;
  u16*   XN2b = B2(NT);
  u16*   H1b  = B2((size_t)TOK * 3072);
  float* M1f  = F4(NT);
  u16*   MA1b = B2((size_t)TOK * 192);
  float* MAf  = F4(NT);

  ln_kernel<<<dim3(TOK), 256, 0, stream>>>(T2, g2, bb2, XN2b);
  GEMM(XN2b, fc1_wb, fc1_b, nullptr, H1b, 3072, 768, 1);
  GEMM(H1b,  fc2_wb, fc2_b, M1f, nullptr, 768, 3072, 0);
  GEMM(XN2b, ma_w1b, ma_b1, nullptr, MA1b, 192, 768, 1);
  GEMM(MA1b, ma_w2b, ma_b2, MAf, nullptr, 768, 192, 0);
  k_add3<<<dim3(EW), 256, 0, stream>>>(outT, T2, M1f, MAf, 0.5f, NT);
}
